// SwinBlock_11330123727266
// MI455X (gfx1250) — matrix operand order repaired and checked by reading, not compiled, not run on hardware
//
#include <hip/hip_runtime.h>
#include <hip/hip_bf16.h>
#include <math.h>

// ---------------------------------------------------------------------------
// Swin block for MI455X (gfx1250, wave32, WMMA 16x16x32 f16).
// GEMMs: block = 8 m-tiles x 6 n-tiles; weights async-staged to LDS
// (global_load_async_to_lds_b128 / ASYNCcnt), B fragments via ds_load_b128.
// ---------------------------------------------------------------------------

typedef _Float16 half_t;
typedef __attribute__((ext_vector_type(16))) _Float16 v16h;
typedef __attribute__((ext_vector_type(8)))  _Float16 v8h;
typedef __attribute__((ext_vector_type(8)))  float    v8f;

union H16 { v16h v; v8h h[2]; };

#define SIZE_  56
#define SHIFT_ 3
#define NWIN   4096      // 64 batch * 64 windows
#define NPAD   64        // 49 tokens padded to 64
#define CCH    96
#define ROWS   200704    // 64 * 3136

__device__ __forceinline__ v8f wmma32(v16h a, v16h b, v8f c) {
    return __builtin_amdgcn_wmma_f32_16x16x32_f16(
        /*neg_a=*/false, a, /*neg_b=*/false, b,
        /*c_mod=*/(short)0, c, /*reuse_a=*/false, /*reuse_b=*/false);
}

// A fragment: 16x32 f16 row-major. lane -> row=lane&15, K=half*16..+15 in v0..7
// (contiguous k order, matching the packed B fragments).
__device__ __forceinline__ v16h load_afrag(const half_t* arow) {
    H16 u;
    u.h[0] = *reinterpret_cast<const v8h*>(arow);
    u.h[1] = *reinterpret_cast<const v8h*>(arow + 8);
    return u.v;
}
// B fragment from 16 contiguous f16 for this lane (packed weights / k rows).
__device__ __forceinline__ v16h load_bfrag(const half_t* bp) {
    H16 u;
    u.h[0] = *reinterpret_cast<const v8h*>(bp);
    u.h[1] = *reinterpret_cast<const v8h*>(bp + 8);
    return u.v;
}

// Async-stage `bytes` (16B multiple) from global into LDS; all 256 threads
// issue global_load_async_to_lds_b128, wait own ASYNCcnt, then barrier.
__device__ __forceinline__ void stage_async(const half_t* __restrict__ gsrc,
                                            half_t* sdst, int bytes, int tid) {
    int nchunks = bytes >> 4;
    unsigned lbase = (unsigned)(size_t)sdst;
    for (int c = tid; c < nchunks; c += 256) {
        unsigned loff = lbase + (unsigned)(c << 4);
        unsigned long long ga = (unsigned long long)((const char*)gsrc + ((size_t)c << 4));
        asm volatile("global_load_async_to_lds_b128 %0, %1, off"
                     :: "v"(loff), "v"(ga) : "memory");
    }
    asm volatile("s_wait_asynccnt 0x0" ::: "memory");
    __syncthreads();
}

// Multi-tile WMMA: one A-fragment feeds NB n-tile accumulators per k-step.
// wlds: LDS panel laid out [nn][ks][lane][16].
template<int KSTEPS, int NB>
__device__ __forceinline__ void tile_mm_multi(const half_t* __restrict__ A, int lda,
                                              const half_t* wlds, int lane, v8f* acc) {
    const int r = lane & 15, hs = lane >> 4;
    const half_t* arow = A + (size_t)r * lda + hs * 16;
#pragma unroll
    for (int ks = 0; ks < KSTEPS; ++ks) {
        __builtin_prefetch(arow + 32, 0, 1);             // global_prefetch_b8
        v16h a = load_afrag(arow);
        arow += 32;
#pragma unroll
        for (int nn = 0; nn < NB; ++nn) {
            v16h b = load_bfrag(wlds + ((size_t)nn * KSTEPS + ks) * 512 + lane * 16);
            acc[nn] = wmma32(a, b, acc[nn]);
        }
    }
}

// ---------------------------------------------------------------------------
// pack f32 weight [K x N] row-major into per-lane B-fragment f16 layout (contiguous k order).
// packed[((nt*ksteps+ks)*32+lane)*16+j] = W[(ks*32+(lane>>4)*16+j)*N + nt*16+(lane&15)]
// ---------------------------------------------------------------------------
__global__ void pack_w_kernel(const float* __restrict__ W, half_t* __restrict__ dst,
                              int N, int ksteps, int total) {
    int p = blockIdx.x * blockDim.x + threadIdx.x;
    if (p >= total) return;
    int j    = p & 15;
    int lane = (p >> 4) & 31;
    int t    = p >> 9;
    int ks   = t % ksteps;
    int nt   = t / ksteps;
    int col  = nt * 16 + (lane & 15);
    int k    = ks * 32 + (lane >> 4) * 16 + j;
    dst[p] = (half_t)W[(size_t)k * N + col];
}

// ---------------------------------------------------------------------------
// combined rel-pos bias + shift mask [64 win][3 head][49 q][64 k]; pad cols -1e9
// ---------------------------------------------------------------------------
__global__ void comb_kernel(const float* __restrict__ bias_table, float* __restrict__ comb) {
    int p = blockIdx.x * blockDim.x + threadIdx.x;
    if (p >= 64 * 3 * 49 * 64) return;
    int j = p & 63;
    int t = p >> 6;
    int i = t % 49;  t /= 49;
    int head = t % 3;
    int wid  = t / 3;
    float val;
    if (j >= 49) {
        val = -1e9f;
    } else {
        int ih = i / 7, iw = i % 7, jh = j / 7, jw = j % 7;
        int idx = (ih - jh + 6) * 13 + (iw - jw + 6);
        float bb = bias_table[idx * 3 + head];
        int wr = wid >> 3, wc = wid & 7;
        auto lab = [](int c) { return c < 49 ? 0 : (c < 53 ? 1 : 2); };
        int li = lab(wr * 7 + ih) * 3 + lab(wc * 7 + iw);
        int lj = lab(wr * 7 + jh) * 3 + lab(wc * 7 + jw);
        val = bb + (li == lj ? 1.0f : 0.0f) - 101.0f;
    }
    comb[p] = val;
}

// ---------------------------------------------------------------------------
// LN1 + cyclic shift + window partition -> h_win f16 [4096][64][96]; pad rows 0
// ---------------------------------------------------------------------------
__global__ void ln1_window_kernel(const float* __restrict__ x,
                                  const float* __restrict__ g,
                                  const float* __restrict__ bt,
                                  half_t* __restrict__ hwin) {
    int win = blockIdx.x;
    int wave = threadIdx.x >> 5, lane = threadIdx.x & 31;
    int bidx = win >> 6, wid = win & 63, wr = wid >> 3, wc = wid & 7;
    for (int n = wave; n < NPAD; n += 8) {
        half_t* orow = hwin + ((size_t)win * NPAD + n) * CCH;
        if (n >= 49) {
            orow[lane] = (half_t)0.f; orow[lane + 32] = (half_t)0.f; orow[lane + 64] = (half_t)0.f;
            continue;
        }
        int ir = n / 7, ic = n % 7;
        int hsrc = (wr * 7 + ir + SHIFT_) % SIZE_;
        int wsrc = (wc * 7 + ic + SHIFT_) % SIZE_;
        const float* xr = x + ((size_t)bidx * (SIZE_ * SIZE_) + hsrc * SIZE_ + wsrc) * CCH;
        float v0 = xr[lane], v1 = xr[lane + 32], v2 = xr[lane + 64];
        float s = v0 + v1 + v2;
        for (int off = 16; off; off >>= 1) s += __shfl_xor(s, off, 32);
        float mean = s * (1.0f / 96.0f);
        float d0 = v0 - mean, d1 = v1 - mean, d2 = v2 - mean;
        float q = d0 * d0 + d1 * d1 + d2 * d2;
        for (int off = 16; off; off >>= 1) q += __shfl_xor(q, off, 32);
        float inv = rsqrtf(q * (1.0f / 96.0f) + 1e-5f);
        orow[lane]      = (half_t)(d0 * inv * g[lane]      + bt[lane]);
        orow[lane + 32] = (half_t)(d1 * inv * g[lane + 32] + bt[lane + 32]);
        orow[lane + 64] = (half_t)(d2 * inv * g[lane + 64] + bt[lane + 64]);
    }
}

// Plain row LN (LN2): one wave per 96-wide row, f16 out.
__global__ void ln_rows_kernel(const float* __restrict__ in,
                               const float* __restrict__ g,
                               const float* __restrict__ bt,
                               half_t* __restrict__ out, int rows) {
    int row = (blockIdx.x * blockDim.x + threadIdx.x) >> 5;
    int lane = threadIdx.x & 31;
    if (row >= rows) return;
    const float* xr = in + (size_t)row * CCH;
    float v0 = xr[lane], v1 = xr[lane + 32], v2 = xr[lane + 64];
    float s = v0 + v1 + v2;
    for (int off = 16; off; off >>= 1) s += __shfl_xor(s, off, 32);
    float mean = s * (1.0f / 96.0f);
    float d0 = v0 - mean, d1 = v1 - mean, d2 = v2 - mean;
    float q = d0 * d0 + d1 * d1 + d2 * d2;
    for (int off = 16; off; off >>= 1) q += __shfl_xor(q, off, 32);
    float inv = rsqrtf(q * (1.0f / 96.0f) + 1e-5f);
    half_t* orow = out + (size_t)row * CCH;
    orow[lane]      = (half_t)(d0 * inv * g[lane]      + bt[lane]);
    orow[lane + 32] = (half_t)(d1 * inv * g[lane + 32] + bt[lane + 32]);
    orow[lane + 64] = (half_t)(d2 * inv * g[lane + 64] + bt[lane + 64]);
}

// ---------------------------------------------------------------------------
// QKV GEMM: block = 8 m-tiles x one group of 6 n-tiles (group 0=q, 1=k, 2=v).
// grid = 2048 m-blocks * 3 groups. Weights (18KB) async-staged to LDS.
// ---------------------------------------------------------------------------
__global__ void qkv_kernel(const half_t* __restrict__ hwin, const half_t* __restrict__ wp,
                           const float* __restrict__ qkvb,
                           half_t* __restrict__ qb, half_t* __restrict__ kb,
                           half_t* __restrict__ vpk) {
    __shared__ __align__(16) half_t smw[6 * 3 * 512];
    int group = blockIdx.x % 3, mb = blockIdx.x / 3;
    int tid = threadIdx.x, wave = tid >> 5, lane = tid & 31;
    stage_async(wp + (size_t)group * 6 * 3 * 512, smw, 6 * 3 * 512 * 2, tid);

    int mtile = mb * 8 + wave;
    int win = mtile >> 2, tm = mtile & 3;
    const half_t* A = hwin + ((size_t)win * NPAD + tm * 16) * CCH;
    v8f acc[6] = {};
    tile_mm_multi<3, 6>(A, CCH, smw, lane, acc);

    int r = lane & 15, hs = lane >> 4;
#pragma unroll
    for (int nn = 0; nn < 6; ++nn) {
        int c = (group * 6 + nn) * 16 + r;          // 0..287, group-uniform kind
        float bias = qkvb[c];
        int head = (c % 96) / 32, d = c % 32;
        size_t base = ((size_t)win * 3 + head) * NPAD;
#pragma unroll
        for (int vv = 0; vv < 8; ++vv) {
            int n = tm * 16 + vv + 8 * hs;
            float val = acc[nn][vv] + bias;
            if (group == 0) {
                qb[(base + n) * 32 + d] = (half_t)(val * 0.17677669529663687f);
            } else if (group == 1) {
                kb[(base + n) * 32 + d] = (half_t)val;
            } else {
                int ntv = d >> 4, kst = n >> 5, kl = n & 31;
                int lv = (kl >> 4) * 16 + (d & 15);
                vpk[base * 32 + (size_t)(ntv * 2 + kst) * 512 + lv * 16 + (kl & 15)] = (half_t)val;
            }
        }
    }
}

// ---------------------------------------------------------------------------
// windowed attention: block=128 (4 waves), one (window,head) per block.
// QK^T (4 WMMA, K=hd=32) -> softmax in C-frag layout -> LDS -> A@V (4 WMMA).
// ---------------------------------------------------------------------------
__global__ void attn_kernel(const half_t* __restrict__ qb, const half_t* __restrict__ kb,
                            const half_t* __restrict__ vpk, const float* __restrict__ comb,
                            half_t* __restrict__ aout) {
    __shared__ __align__(16) half_t sm[4 * 16 * 64];
    int job  = blockIdx.x;               // 4096*3
    int win  = job / 3, head = job % 3;
    int wave = threadIdx.x >> 5, lane = threadIdx.x & 31;
    int tm = wave;
    int r = lane & 15, hs = lane >> 4;
    size_t base = ((size_t)win * 3 + head) * NPAD;

    v16h qa = load_afrag(qb + (base + tm * 16 + r) * 32 + hs * 16);

    v8f sc[4];
#pragma unroll
    for (int ct = 0; ct < 4; ++ct) {
        v16h bf = load_bfrag(kb + (base + ct * 16 + r) * 32 + hs * 16);
        v8f z = {};
        sc[ct] = wmma32(qa, bf, z);
    }

    int wid = win & 63;
    const float* cb = comb + ((size_t)(wid * 3 + head)) * 49 * 64;
    half_t* lds = sm + wave * 1024;
#pragma unroll
    for (int vv = 0; vv < 8; ++vv) {
        int rr = tm * 16 + vv + 8 * hs;
        float s0 = sc[0][vv], s1 = sc[1][vv], s2 = sc[2][vv], s3 = sc[3][vv];
        if (rr < 49) {
            const float* cr = cb + (size_t)rr * 64;
            s0 += cr[r]; s1 += cr[16 + r]; s2 += cr[32 + r]; s3 += cr[48 + r];
        }
        float m = fmaxf(fmaxf(s0, s1), fmaxf(s2, s3));
        for (int off = 8; off; off >>= 1) m = fmaxf(m, __shfl_xor(m, off, 32));
        float e0 = __expf(s0 - m), e1 = __expf(s1 - m), e2 = __expf(s2 - m), e3 = __expf(s3 - m);
        float su = e0 + e1 + e2 + e3;
        for (int off = 8; off; off >>= 1) su += __shfl_xor(su, off, 32);
        float inv = 1.0f / su;
        int lr = vv + 8 * hs;
        lds[lr * 64 +      r] = (half_t)(e0 * inv);
        lds[lr * 64 + 16 + r] = (half_t)(e1 * inv);
        lds[lr * 64 + 32 + r] = (half_t)(e2 * inv);
        lds[lr * 64 + 48 + r] = (half_t)(e3 * inv);
    }

    const half_t* vb = vpk + base * 32;
#pragma unroll
    for (int ntv = 0; ntv < 2; ++ntv) {
        v8f acc = {};
#pragma unroll
        for (int kst = 0; kst < 2; ++kst) {
            v16h a2 = load_afrag(lds + r * 64 + kst * 32 + hs * 16);
            v16h b2 = load_bfrag(vb + (size_t)(ntv * 2 + kst) * 512 + lane * 16);
            acc = wmma32(a2, b2, acc);
        }
#pragma unroll
        for (int vv = 0; vv < 8; ++vv) {
            int n = tm * 16 + vv + 8 * hs;
            aout[((size_t)win * NPAD + n) * CCH + head * 32 + ntv * 16 + r] = (half_t)acc[vv];
        }
    }
}

// ---------------------------------------------------------------------------
// proj GEMM (all 6 n-tiles per wave) + window-reverse + un-shift + residual
// ---------------------------------------------------------------------------
__global__ void proj_kernel(const half_t* __restrict__ aout, const half_t* __restrict__ wp,
                            const float* __restrict__ pb, const float* __restrict__ x,
                            float* __restrict__ out1) {
    __shared__ __align__(16) half_t smw[6 * 3 * 512];
    int tid = threadIdx.x, wave = tid >> 5, lane = tid & 31;
    stage_async(wp, smw, 6 * 3 * 512 * 2, tid);

    int mtile = blockIdx.x * 8 + wave;
    int win = mtile >> 2, tm = mtile & 3;
    const half_t* A = aout + ((size_t)win * NPAD + tm * 16) * CCH;
    v8f acc[6] = {};
    tile_mm_multi<3, 6>(A, CCH, smw, lane, acc);

    int r = lane & 15, hs = lane >> 4;
    int bidx = win >> 6, wid = win & 63, wr = wid >> 3, wc = wid & 7;
#pragma unroll
    for (int nn = 0; nn < 6; ++nn) {
        int c = nn * 16 + r;
        float bias = pb[c];
#pragma unroll
        for (int vv = 0; vv < 8; ++vv) {
            int n = tm * 16 + vv + 8 * hs;
            if (n >= 49) continue;
            int ir = n / 7, ic = n % 7;
            int hh = (wr * 7 + ir + SHIFT_) % SIZE_;
            int ww = (wc * 7 + ic + SHIFT_) % SIZE_;
            size_t o = ((size_t)bidx * (SIZE_ * SIZE_) + hh * SIZE_ + ww) * CCH + c;
            out1[o] = x[o] + acc[nn][vv] + bias;
        }
    }
}

// ---------------------------------------------------------------------------
// fc1 GEMM + tanh-GELU -> f16 [ROWS x 384]; 4 n-groups of 6 tiles
// ---------------------------------------------------------------------------
__global__ void fc1_kernel(const half_t* __restrict__ A, const half_t* __restrict__ wp,
                           const float* __restrict__ bias, half_t* __restrict__ out) {
    __shared__ __align__(16) half_t smw[6 * 3 * 512];
    int group = blockIdx.x & 3, mb = blockIdx.x >> 2;
    int tid = threadIdx.x, wave = tid >> 5, lane = tid & 31;
    stage_async(wp + (size_t)group * 6 * 3 * 512, smw, 6 * 3 * 512 * 2, tid);

    int mtile = mb * 8 + wave;
    v8f acc[6] = {};
    tile_mm_multi<3, 6>(A + (size_t)mtile * 16 * CCH, CCH, smw, lane, acc);

    int r = lane & 15, hs = lane >> 4;
#pragma unroll
    for (int nn = 0; nn < 6; ++nn) {
        int c = (group * 6 + nn) * 16 + r;
        float bb = bias[c];
#pragma unroll
        for (int vv = 0; vv < 8; ++vv) {
            float xv = acc[nn][vv] + bb;
            float u = 0.7978845608028654f * (xv + 0.044715f * xv * xv * xv);
            float gl = 0.5f * xv * (1.0f + tanhf(u));
            out[((size_t)(mtile * 16 + vv + 8 * hs)) * 384 + c] = (half_t)gl;
        }
    }
}

// ---------------------------------------------------------------------------
// fc2 GEMM (K=384 -> 12 k-steps, all 6 n-tiles) + residual -> d_out (f32)
// ---------------------------------------------------------------------------
__global__ void fc2_kernel(const half_t* __restrict__ A, const half_t* __restrict__ wp,
                           const float* __restrict__ bias, const float* __restrict__ res,
                           float* __restrict__ out) {
    __shared__ __align__(16) half_t smw[6 * 12 * 512];
    int tid = threadIdx.x, wave = tid >> 5, lane = tid & 31;
    stage_async(wp, smw, 6 * 12 * 512 * 2, tid);

    int mtile = blockIdx.x * 8 + wave;
    v8f acc[6] = {};
    tile_mm_multi<12, 6>(A + (size_t)mtile * 16 * 384, 384, smw, lane, acc);

    int r = lane & 15, hs = lane >> 4;
#pragma unroll
    for (int nn = 0; nn < 6; ++nn) {
        int c = nn * 16 + r;
        float bb = bias[c];
#pragma unroll
        for (int vv = 0; vv < 8; ++vv) {
            size_t o = ((size_t)(mtile * 16 + vv + 8 * hs)) * CCH + c;
            out[o] = res[o] + acc[nn][vv] + bb;
        }
    }
}

// ---------------------------------------------------------------------------
// Workspace layout (bytes). Peak ~318 MB with region reuse.
// ---------------------------------------------------------------------------
#define OFF_WP_QKV  0ull
#define OFF_WP_PROJ 55296ull
#define OFF_WP_FC1  73728ull
#define OFF_WP_FC2  147456ull
#define OFF_COMB    221184ull          // 64*3*49*64*4 = 2408448 bytes
#define OFF_A       (4ull << 20)       // region A (192 MB), reused
#define SZ_HWIN     50331648ull        // 4096*64*96*2
#define OFF_Q       (OFF_A + SZ_HWIN)
#define OFF_K       (OFF_A + 2 * SZ_HWIN)
#define OFF_V       (OFF_A + 3 * SZ_HWIN)
#define OFF_LN2     OFF_A              // reuse (h_win dead)
#define OFF_FC1O    (OFF_A + (40ull << 20))  // reuse (qkv dead), 154.1 MB
#define OFF_AOUT    (OFF_A + 4 * SZ_HWIN)    // 50.3 MB
#define OFF_OUT1    (OFF_AOUT + SZ_HWIN)     // 77.1 MB

extern "C" void kernel_launch(void* const* d_in, const int* in_sizes, int n_in,
                              void* d_out, int out_size, void* d_ws, size_t ws_size,
                              hipStream_t stream) {
    const float* x        = (const float*)d_in[0];
    const float* n1g      = (const float*)d_in[1];
    const float* n1b      = (const float*)d_in[2];
    const float* qkv_w    = (const float*)d_in[3];
    const float* qkv_b    = (const float*)d_in[4];
    const float* rel_bias = (const float*)d_in[5];
    const float* proj_w   = (const float*)d_in[6];
    const float* proj_b   = (const float*)d_in[7];
    const float* n2g      = (const float*)d_in[8];
    const float* n2b      = (const float*)d_in[9];
    const float* fc1_w    = (const float*)d_in[10];
    const float* fc1_b    = (const float*)d_in[11];
    const float* fc2_w    = (const float*)d_in[12];
    const float* fc2_b    = (const float*)d_in[13];
    (void)in_sizes; (void)n_in; (void)out_size; (void)ws_size;

    char* ws = (char*)d_ws;
    half_t* wp_qkv  = (half_t*)(ws + OFF_WP_QKV);
    half_t* wp_proj = (half_t*)(ws + OFF_WP_PROJ);
    half_t* wp_fc1  = (half_t*)(ws + OFF_WP_FC1);
    half_t* wp_fc2  = (half_t*)(ws + OFF_WP_FC2);
    float*  comb    = (float*)(ws + OFF_COMB);
    half_t* hwin    = (half_t*)(ws + OFF_A);
    half_t* qb      = (half_t*)(ws + OFF_Q);
    half_t* kb      = (half_t*)(ws + OFF_K);
    half_t* vpk     = (half_t*)(ws + OFF_V);
    half_t* ln2b    = (half_t*)(ws + OFF_LN2);
    half_t* fc1o    = (half_t*)(ws + OFF_FC1O);
    half_t* aoutb   = (half_t*)(ws + OFF_AOUT);
    float*  out1    = (float*)(ws + OFF_OUT1);
    float*  outp    = (float*)d_out;

    // --- weight packing + bias/mask table ---
    pack_w_kernel<<<(96 * 288 + 255) / 256, 256, 0, stream>>>(qkv_w,  wp_qkv,  288, 3,  96 * 288);
    pack_w_kernel<<<(96 * 96  + 255) / 256, 256, 0, stream>>>(proj_w, wp_proj, 96,  3,  96 * 96);
    pack_w_kernel<<<(96 * 384 + 255) / 256, 256, 0, stream>>>(fc1_w,  wp_fc1,  384, 3,  96 * 384);
    pack_w_kernel<<<(384 * 96 + 255) / 256, 256, 0, stream>>>(fc2_w,  wp_fc2,  96,  12, 384 * 96);
    comb_kernel<<<(64 * 3 * 49 * 64 + 255) / 256, 256, 0, stream>>>(rel_bias, comb);

    // --- LN1 + shift + window partition ---
    ln1_window_kernel<<<NWIN, 256, 0, stream>>>(x, n1g, n1b, hwin);

    // --- QKV GEMM: 2048 m-blocks x 3 n-groups ---
    qkv_kernel<<<(NWIN * 4 / 8) * 3, 256, 0, stream>>>(hwin, wp_qkv, qkv_b, qb, kb, vpk);

    // --- windowed attention ---
    attn_kernel<<<NWIN * 3, 128, 0, stream>>>(qb, kb, vpk, comb, aoutb);

    // --- proj + window reverse + un-shift + residual ---
    proj_kernel<<<NWIN * 4 / 8, 256, 0, stream>>>(aoutb, wp_proj, proj_b, x, out1);

    // --- LN2 ---
    ln_rows_kernel<<<ROWS / 8, 256, 0, stream>>>(out1, n2g, n2b, ln2b, ROWS);

    // --- MLP ---
    fc1_kernel<<<(ROWS / 16 / 8) * 4, 256, 0, stream>>>(ln2b, wp_fc1, fc1_b, fc1o);
    fc2_kernel<<<ROWS / 16 / 8, 256, 0, stream>>>(fc1o, wp_fc2, fc2_b, out1, outp);
}
